// MyGenerator_25829933318369
// MI455X (gfx1250) — matrix operand order repaired and checked by reading, not compiled, not run on hardware
//
#include <hip/hip_runtime.h>
#include <math.h>

typedef __attribute__((ext_vector_type(16))) _Float16 v16h;
typedef __attribute__((ext_vector_type(8)))  _Float16 v8h;
typedef __attribute__((ext_vector_type(8)))  float    v8f;

static __device__ __forceinline__ v16h cat8(v8h lo, v8h hi) {
  return __builtin_shufflevector(lo, hi, 0, 1, 2, 3, 4, 5, 6, 7,
                                 8, 9, 10, 11, 12, 13, 14, 15);
}

// ---------------------------------------------------------------------------
// Implicit-GEMM conv via WMMA f16 (accum f32).
// MODE 0: KDIMxKDIM conv, stride 1, reflect pad KDIM/2  (reflect_conv7)
// MODE 1: 3x3 conv, stride 2, zero pad 1                (conv_block)
// MODE 2: 3x3 transposed conv, stride 2 (2x upsample)   (deconv_block)
//         optional fused skip-add on the INPUT (x + skip)
// GEMM view: M = B*Ho*Wo pixels, N = Cout, K = Cin*KDIM*KDIM
// ---------------------------------------------------------------------------
template<int MODE, int KDIM>
__device__ __forceinline__ float gatherA(const float* __restrict__ x,
                                         const float* __restrict__ skip,
                                         int b, int oh, int ow, int k,
                                         int Cin, int Hi, int Wi, int K) {
  if (k >= K) return 0.f;
  int ci = k / (KDIM * KDIM);
  int rem = k - ci * (KDIM * KDIM);
  int ki = rem / KDIM;
  int kj = rem - ki * KDIM;
  int ih, iw;
  if (MODE == 0) {                      // reflect pad
    ih = oh - KDIM / 2 + ki; iw = ow - KDIM / 2 + kj;
    ih = ih < 0 ? -ih : (ih >= Hi ? 2 * Hi - 2 - ih : ih);
    iw = iw < 0 ? -iw : (iw >= Wi ? 2 * Wi - 2 - iw : iw);
  } else if (MODE == 1) {               // stride-2, zero pad 1
    ih = oh * 2 - 1 + ki; iw = ow * 2 - 1 + kj;
    if ((unsigned)ih >= (unsigned)Hi || (unsigned)iw >= (unsigned)Wi) return 0.f;
  } else {                              // transposed stride-2
    int th = oh + 1 - ki, tw = ow + 1 - kj;
    if ((th & 1) | (tw & 1)) return 0.f;
    ih = th >> 1; iw = tw >> 1;
    if ((unsigned)ih >= (unsigned)Hi || (unsigned)iw >= (unsigned)Wi) return 0.f;
  }
  size_t idx = (((size_t)b * Cin + ci) * Hi + ih) * Wi + iw;
  float v = x[idx];
  if (MODE == 2 && skip) v += skip[idx];
  return v;
}

template<int MODE, int KDIM>
__global__ __launch_bounds__(128)
void conv_wmma(const float* __restrict__ x, const float* __restrict__ skip,
               const float* __restrict__ w, const float* __restrict__ bias,
               float* __restrict__ y, int Bn, int Cin, int Hi, int Wi,
               int Cout, int Ho, int Wo, int act) {
  const int K = Cin * KDIM * KDIM;
  const int M = Bn * Ho * Wo;
  const int mt = (M + 15) >> 4, nt = (Cout + 15) >> 4;
  int wid = blockIdx.x * (blockDim.x >> 5) + (threadIdx.x >> 5);
  if (wid >= mt * nt) return;                 // wave-uniform exit
  int tn = wid % nt, tm = wid / nt;
  int lane = threadIdx.x & 31;
  int half = lane >> 4;
  int l16  = lane & 15;
  int m0 = tm << 4, n0 = tn << 4;

  // pixel decode hoisted out of K loop (clamped; padded rows never stored)
  int am = m0 + l16;
  int amc = am < M ? am : 0;
  int ow = amc % Wo; int t = amc / Wo; int oh = t % Ho; int bimg = t / Ho;

  int co = n0 + l16;
  bool cok = co < Cout;
  const float* wrow = w + (size_t)(cok ? co : 0) * K;
  const bool wvec = ((K & 3) == 0);           // float4 path provably 16B aligned

  v8f acc = {};
  for (int k0 = 0; k0 < K; k0 += 32) {
    v16h a, bm;
#pragma unroll
    for (int e = 0; e < 16; ++e) {
      // A 16x32 f16 layout: lane m=lane&15; k = k0 + half*16 + e  (contiguous)
      int ka = k0 + (half << 4) + e;
      a[e] = (_Float16)gatherA<MODE, KDIM>(x, skip, bimg, oh, ow, ka, Cin, Hi, Wi, K);
    }
    // B 32x16 f16 layout: lane n=lane&15; k = k0 + half*16 + e  (contiguous)
    int kb0 = k0 + (half << 4);
    if (wvec && kb0 + 16 <= K) {
      const float4* wv = (const float4*)(wrow + kb0);
      float4 w0 = wv[0], w1 = wv[1], w2 = wv[2], w3 = wv[3];
      bm[0]  = (_Float16)w0.x; bm[1]  = (_Float16)w0.y;
      bm[2]  = (_Float16)w0.z; bm[3]  = (_Float16)w0.w;
      bm[4]  = (_Float16)w1.x; bm[5]  = (_Float16)w1.y;
      bm[6]  = (_Float16)w1.z; bm[7]  = (_Float16)w1.w;
      bm[8]  = (_Float16)w2.x; bm[9]  = (_Float16)w2.y;
      bm[10] = (_Float16)w2.z; bm[11] = (_Float16)w2.w;
      bm[12] = (_Float16)w3.x; bm[13] = (_Float16)w3.y;
      bm[14] = (_Float16)w3.z; bm[15] = (_Float16)w3.w;
    } else {
#pragma unroll
      for (int e = 0; e < 16; ++e) {
        int kb = kb0 + e;
        bm[e] = (_Float16)((kb < K) ? wrow[kb] : 0.f);
      }
    }
    acc = __builtin_amdgcn_wmma_f32_16x16x32_f16(false, a, false, bm,
                                                 (short)0, acc, false, false);
  }
  if (!cok) return;
  float bb = bias ? bias[co] : 0.f;
#pragma unroll
  for (int r = 0; r < 8; ++r) {
    int m = m0 + (half << 3) + r;            // C layout: m=(lane>=16?8:0)+vgpr, n=lane&15
    if (m >= M) continue;
    float v = acc[r] + bb;
    if (act == 1)      v = v > 0.f ? v : 0.2f * v;   // LeakyReLU(0.2)
    else if (act == 2) v = tanhf(v);
    int ow2 = m % Wo; int t2 = m / Wo; int oh2 = t2 % Ho; int b2 = t2 / Ho;
    y[(((size_t)b2 * Cout + co) * Ho + oh2) * Wo + ow2] = v;
  }
}

// ---------------------------------------------------------------------------
// get_color_map helpers
// ---------------------------------------------------------------------------
__global__ void masked_mean_kernel(const float* __restrict__ featT,
                                   const float* __restrict__ featS,
                                   const int* __restrict__ tmask,
                                   const int* __restrict__ smask,
                                   float* __restrict__ mu) {
  int idx = blockIdx.x;                       // 0..11
  int which = idx & 1; int reg = (idx >> 1) % 3; int b = idx / 6;
  const float* f = (which ? featS : featT) + (size_t)b * 64 * 4096;
  const int*   m = (which ? smask : tmask) + ((size_t)b * 4 + reg + 1) * 4096;
  int c = threadIdx.x;                        // 64 threads
  const float* fc = f + (size_t)c * 4096;
  float s = 0.f, ms = 0.f;
  for (int i = 0; i < 4096; ++i) {
    float mm = (m[i] > 0) ? 1.f : 0.f;
    s += fc[i] * mm; ms += mm;
  }
  mu[idx * 64 + c] = s / ms;
}

// nrm[which][b][reg][hw][64] f16 : per-pixel L2-normalized centered features
__global__ void normalize_kernel(const float* __restrict__ featT,
                                 const float* __restrict__ featS,
                                 const float* __restrict__ mu,
                                 _Float16* __restrict__ out) {
  int idx = blockIdx.y;
  int which = idx & 1; int reg = (idx >> 1) % 3; int b = idx / 6;
  int hw = blockIdx.x * blockDim.x + threadIdx.x;
  const float* f  = (which ? featS : featT) + (size_t)b * 64 * 4096 + hw;
  const float* mc = mu + idx * 64;
  float bar[64]; float ss = 0.f;
#pragma unroll 8
  for (int c = 0; c < 64; ++c) { float v = f[(size_t)c * 4096] - mc[c]; bar[c] = v; ss += v * v; }
  float inv = rsqrtf(fmaxf(ss, 1e-12f));
  _Float16* o = out + (((size_t)which * 6 + b * 3 + reg) * 4096 + hw) * 64;
#pragma unroll 8
  for (int c = 0; c < 64; ++c) o[c] = (_Float16)(bar[c] * inv);
}

// ---------------------------------------------------------------------------
// Flash-style masked cross attention.  One wave owns 16 target rows, streams
// 4096 source pixels as 256 tiles of 16.  Tiles are double-buffered in LDS
// via async global->LDS copies (ASYNCcnt pipelined one tile ahead).
// Scores are cosine correlations of L2-normalized vectors => s <= 1, so the
// softmax uses a FIXED shift of 1.0 (shift-invariant, overflow-free): no
// online max, no rescale; cross-lane reductions deferred to a single epilogue.
// ---------------------------------------------------------------------------
#define ASYNC_TILE_COPY(LDSA, GA)                                         \
  asm volatile(                                                           \
      "global_load_async_to_lds_b128 %0, %1, off\n\t"                     \
      "global_load_async_to_lds_b128 %0, %1, off offset:512\n\t"          \
      "global_load_async_to_lds_b128 %0, %1, off offset:1024\n\t"         \
      "global_load_async_to_lds_b128 %0, %1, off offset:1536"             \
      :: "v"(LDSA), "v"(GA) : "memory")

__global__ __launch_bounds__(32)
void attn_kernel(const _Float16* __restrict__ nrm, const int* __restrict__ smask,
                 const float* __restrict__ scol, float* __restrict__ cols) {
  __shared__ _Float16 stile[2][1024];         // 2 x (16 px x 64 ch) f16
  int breg = blockIdx.y; int b = breg / 3, reg = breg % 3;
  int i0 = blockIdx.x << 4;
  int lane = threadIdx.x; int half = lane >> 4; int l16 = lane & 15;
  const _Float16* tq = nrm + (size_t)breg * 4096 * 64;
  const _Float16* sq = nrm + (size_t)(6 + breg) * 4096 * 64;
  const int*   sm = smask + ((size_t)b * 4 + reg + 1) * 4096;
  const float* sc = scol + (size_t)b * 3 * 4096;

  // Q rows -> A operands (two K=32 chunks), per ISA 16-bit A layout
  v16h a0, a1;
  {
    const v8h* rowv = (const v8h*)(tq + (size_t)(i0 + l16) * 64);
    a0 = cat8(rowv[half],     rowv[2 + half]);
    a1 = cat8(rowv[4 + half], rowv[6 + half]);
  }

  float lrow[8], o0[8], o1[8], o2[8];
#pragma unroll
  for (int r = 0; r < 8; ++r) { lrow[r] = 0.f; o0[r] = o1[r] = o2[r] = 0.f; }

  unsigned lds0 = (unsigned)(size_t)&stile[0][0] + (unsigned)lane * 16u;
  unsigned lds1 = (unsigned)(size_t)&stile[1][0] + (unsigned)lane * 16u;
  const char* gbase = (const char*)sq + (size_t)lane * 16;

  ASYNC_TILE_COPY(lds0, gbase);               // prime tile 0 -> buffer 0
  for (int jt = 0; jt < 256; ++jt) {
    int buf = jt & 1;
    {                                         // issue next tile (clamped)
      int jn = jt + 1 < 256 ? jt + 1 : 255;
      const char* g = gbase + (size_t)jn * 2048;
      unsigned l = buf ? lds0 : lds1;
      ASYNC_TILE_COPY(l, g);
    }
    asm volatile("s_wait_asynccnt 0x4" ::: "memory");  // our tile's 4 copies done

    const v8h* lv = (const v8h*)&stile[buf][l16 * 64]; // K/B layout: k = half*8+e (e<8), 16+half*8+(e-8) (e>=8)
    v16h b0 = cat8(lv[half],     lv[2 + half]);
    v16h b1 = cat8(lv[4 + half], lv[6 + half]);
    v8f s = {};
    s = __builtin_amdgcn_wmma_f32_16x16x32_f16(false, a0, false, b0, (short)0, s, false, false);
    s = __builtin_amdgcn_wmma_f32_16x16x32_f16(false, a1, false, b1, (short)0, s, false, false);

    int j = (jt << 4) + l16;
    bool msk = sm[j] > 0;
    float c0 = sc[j], c1 = sc[4096 + j], c2 = sc[8192 + j];
#pragma unroll
    for (int r = 0; r < 8; ++r) {
      float p = msk ? __expf(s[r] - 1.0f) : 0.f;   // fixed-shift softmax numerator
      lrow[r] += p;
      o0[r] = fmaf(p, c0, o0[r]);
      o1[r] = fmaf(p, c1, o1[r]);
      o2[r] = fmaf(p, c2, o2[r]);
    }
  }

  // single deferred reduction over the 16 columns of each half-wave
#pragma unroll
  for (int r = 0; r < 8; ++r) {
#pragma unroll
    for (int d = 1; d <= 8; d <<= 1) {
      lrow[r] += __shfl_xor(lrow[r], d, 32);
      o0[r]   += __shfl_xor(o0[r], d, 32);
      o1[r]   += __shfl_xor(o1[r], d, 32);
      o2[r]   += __shfl_xor(o2[r], d, 32);
    }
  }
  if (l16 == 0) {
    float* cb = cols + (size_t)breg * 3 * 4096;
#pragma unroll
    for (int r = 0; r < 8; ++r) {
      int i = i0 + (half << 3) + r;
      float inv = lrow[r] > 0.f ? 1.f / lrow[r] : 0.f;
      cb[i] = o0[r] * inv; cb[4096 + i] = o1[r] * inv; cb[8192 + i] = o2[r] * inv;
    }
  }
}

// Sequential-overwrite canvas compose (later region wins), writes color_map
__global__ void canvas_kernel(const float* __restrict__ cols,
                              const int* __restrict__ tmask,
                              float* __restrict__ cmap) {
  int i = blockIdx.x * 256 + threadIdx.x;
  if (i >= 2 * 4096) return;
  int b = i >> 12, hw = i & 4095;
  float v0 = 0.f, v1 = 0.f, v2 = 0.f;
  for (int c = 0; c < 3; ++c) {
    if (tmask[((size_t)b * 4 + c + 1) * 4096 + hw] > 0) {
      const float* cb = cols + (size_t)(b * 3 + c) * 3 * 4096;
      v0 = cb[hw]; v1 = cb[4096 + hw]; v2 = cb[8192 + hw];
    }
  }
  cmap[((size_t)b * 3 + 0) * 4096 + hw] = v0;
  cmap[((size_t)b * 3 + 1) * 4096 + hw] = v1;
  cmap[((size_t)b * 3 + 2) * 4096 + hw] = v2;
}

// concat(target_gray, color_map) -> [B,4,HW]
__global__ void concat_kernel(const float* __restrict__ tgray,
                              const float* __restrict__ cmap,
                              float* __restrict__ uin) {
  int i = blockIdx.x * 256 + threadIdx.x;
  if (i >= 2 * 4 * 4096) return;
  int hw = i & 4095; int c = (i >> 12) & 3; int b = i >> 14;
  uin[i] = (c == 0) ? tgray[(size_t)b * 4096 + hw]
                    : cmap[((size_t)b * 3 + (c - 1)) * 4096 + hw];
}

// ---------------------------------------------------------------------------
// Host orchestration
// ---------------------------------------------------------------------------
struct NetW {
  const float *inpw, *inpb, *d1w, *d1b, *d2w, *d2b, *d3w, *d3b;
  const float *u3w, *u3b, *u2w, *u2b, *u1w, *u1b;
};

static inline int cblocks(int M, int N) {
  return (((M + 15) / 16) * ((N + 15) / 16) + 3) / 4;   // 4 waves / block
}

static void run_backbone(const float* img, int Cin0, const NetW& w,
                         float* bx, float* bx1, float* bx2, float* bx3,
                         float* bu3, float* bu2, float* feat, bool skips,
                         hipStream_t s) {
  conv_wmma<0,7><<<cblocks(2*64*64, 64),128,0,s>>>(img, nullptr, w.inpw, w.inpb, bx, 2, Cin0,64,64, 64,64,64, 0);
  conv_wmma<1,3><<<cblocks(2*32*32,128),128,0,s>>>(bx,  nullptr, w.d1w, w.d1b, bx1, 2,  64,64,64, 128,32,32, 1);
  conv_wmma<1,3><<<cblocks(2*16*16,256),128,0,s>>>(bx1, nullptr, w.d2w, w.d2b, bx2, 2, 128,32,32, 256,16,16, 1);
  conv_wmma<1,3><<<cblocks(2* 8* 8,512),128,0,s>>>(bx2, nullptr, w.d3w, w.d3b, bx3, 2, 256,16,16, 512, 8, 8, 1);
  conv_wmma<2,3><<<cblocks(2*16*16,256),128,0,s>>>(bx3, nullptr, w.u3w, w.u3b, bu3, 2, 512, 8, 8, 256,16,16, 1);
  conv_wmma<2,3><<<cblocks(2*32*32,128),128,0,s>>>(skips ? bx2 : bu3, skips ? bu3 : nullptr,
                                                   w.u2w, w.u2b, bu2, 2, 256,16,16, 128,32,32, 1);
  conv_wmma<2,3><<<cblocks(2*64*64, 64),128,0,s>>>(skips ? bx1 : bu2, skips ? bu2 : nullptr,
                                                   w.u1w, w.u1b, feat, 2, 128,32,32, 64,64,64, 1);
}

extern "C" void kernel_launch(void* const* d_in, const int* in_sizes, int n_in,
                              void* d_out, int out_size, void* d_ws, size_t ws_size,
                              hipStream_t stream) {
  // inputs (pytree order, dict keys sorted):
  // 0 source_color 1 source_gray 2 source_mask 3 target_color 4 target_gray 5 target_mask
  // fpn: 6 d1w 7 d1b 8 d2w 9 d2b 10 d3w 11 d3b 12 inpw 13 inpb 14 u1w 15 u1b 16 u2w 17 u2b 18 u3w 19 u3b
  // unet: 20 d1w 21 d1b 22 d2w 23 d2b 24 d3w 25 d3b 26 inpw 27 inpb 28 outw 29 outb
  //       30 u1w 31 u1b 32 u2w 33 u2b 34 u3w 35 u3b
  const float* scolor = (const float*)d_in[0];
  const float* tgray  = (const float*)d_in[4];
  const int*   smask  = (const int*)d_in[2];
  const float* tcolor = (const float*)d_in[3];
  const int*   tmask  = (const int*)d_in[5];
  NetW fpn = { (const float*)d_in[12], (const float*)d_in[13],
               (const float*)d_in[6],  (const float*)d_in[7],
               (const float*)d_in[8],  (const float*)d_in[9],
               (const float*)d_in[10], (const float*)d_in[11],
               (const float*)d_in[18], (const float*)d_in[19],
               (const float*)d_in[16], (const float*)d_in[17],
               (const float*)d_in[14], (const float*)d_in[15] };
  NetW unet = { (const float*)d_in[26], (const float*)d_in[27],
                (const float*)d_in[20], (const float*)d_in[21],
                (const float*)d_in[22], (const float*)d_in[23],
                (const float*)d_in[24], (const float*)d_in[25],
                (const float*)d_in[34], (const float*)d_in[35],
                (const float*)d_in[32], (const float*)d_in[33],
                (const float*)d_in[30], (const float*)d_in[31] };
  const float* outw = (const float*)d_in[28];
  const float* outb = (const float*)d_in[29];

  float* ws = (float*)d_ws;
  float* buf_x  = ws;                       // 2*64*4096
  float* buf_x1 = buf_x  + 524288;          // 2*128*1024
  float* buf_x2 = buf_x1 + 262144;          // 2*256*256
  float* buf_x3 = buf_x2 + 131072;          // 2*512*64
  float* buf_u3 = buf_x3 + 65536;           // 2*256*256
  float* buf_u2 = buf_u3 + 131072;          // 2*128*1024
  float* featS  = buf_u2 + 262144;          // 2*64*4096
  float* featT  = featS  + 524288;          // 2*64*4096
  float* mu     = featT  + 524288;          // 12*64
  float* cols   = mu     + 768;             // 2*3*3*4096
  float* uin    = cols   + 73728;           // 2*4*4096
  _Float16* nrm = (_Float16*)(uin + 32768); // 2*2*3*4096*64 halves

  float* out_blend = (float*)d_out;         // [2,3,64,64]
  float* out_cmap  = out_blend + 24576;     // [2,3,64,64]

  // FPN on source and target
  run_backbone(scolor, 3, fpn, buf_x, buf_x1, buf_x2, buf_x3, buf_u3, buf_u2, featS, true, stream);
  run_backbone(tcolor, 3, fpn, buf_x, buf_x1, buf_x2, buf_x3, buf_u3, buf_u2, featT, true, stream);

  // color map: normalize -> flash attention (WMMA + async LDS) -> canvas
  masked_mean_kernel<<<12, 64, 0, stream>>>(featT, featS, tmask, smask, mu);
  normalize_kernel<<<dim3(16, 12), 256, 0, stream>>>(featT, featS, mu, nrm);
  attn_kernel<<<dim3(256, 6), 32, 0, stream>>>(nrm, smask, scolor, cols);
  canvas_kernel<<<(8192 + 255) / 256, 256, 0, stream>>>(cols, tmask, out_cmap);

  // UNet blend
  concat_kernel<<<(32768 + 255) / 256, 256, 0, stream>>>(tgray, out_cmap, uin);
  run_backbone(uin, 4, unet, buf_x, buf_x1, buf_x2, buf_x3, buf_u3, buf_u2, buf_x, false, stream);
  conv_wmma<0,7><<<cblocks(2*64*64, 3), 128, 0, stream>>>(buf_x, nullptr, outw, outb,
                                                          out_blend, 2, 64,64,64, 3,64,64, 2);
  (void)in_sizes; (void)n_in; (void)out_size; (void)ws_size;
}
